// LanguageRouterMoeTransformerEncoder_57509612093492
// MI455X (gfx1250) — matrix operand order repaired and checked by reading, not compiled, not run on hardware
//
#include <hip/hip_runtime.h>

// ---------------- model constants ----------------
#define NLAYER 2
#define DIM    1024
#define NHEADS 16
#define DHEAD  64
#define INNER  1024
#define MLPD   4096
#define NEXP   8
#define TOPK   2
#define NB     8
#define SEQ    512
#define LROUT  77
#define CAPV   2
#define EPSV   1e-5f

typedef __bf16 bf16_t;
typedef __attribute__((ext_vector_type(16))) __bf16 v16bf;
typedef __attribute__((ext_vector_type(8)))  __bf16 v8bf;
typedef __attribute__((ext_vector_type(8)))  float  v8f;
typedef __attribute__((ext_vector_type(4)))  int    v4i;

// CDNA5 async global->LDS direct copy (bypasses VGPRs, tracked by ASYNCcnt).
#if defined(__gfx1250__) && __has_builtin(__builtin_amdgcn_global_load_async_to_lds_b128)
#define HAVE_ASYNC_LDS 1
#endif

__device__ inline bf16_t f2bf(float f) {
  unsigned u = __builtin_bit_cast(unsigned, f);
  unsigned r = (u + 0x7FFFu + ((u >> 16) & 1u)) >> 16;
  return __builtin_bit_cast(bf16_t, (unsigned short)r);
}
__device__ inline v8bf zero8() {
  v8bf z;
#pragma unroll
  for (int j = 0; j < 8; ++j) z[j] = __builtin_bit_cast(bf16_t, (unsigned short)0);
  return z;
}
__device__ inline float gelu_f(float x) {
  return 0.5f * x * (1.0f + erff(x * 0.7071067811865475f));
}

#ifdef HAVE_ASYNC_LDS
typedef __attribute__((address_space(1))) v4i gv4i_t;   // global
typedef __attribute__((address_space(3))) v4i lv4i_t;   // LDS
__device__ inline void async_cp16(const bf16_t* g, bf16_t* l) {
  __builtin_amdgcn_global_load_async_to_lds_b128(
      (gv4i_t*)g, (lv4i_t*)l, 0, 0);
}
__device__ inline void wait_async0() {
#if __has_builtin(__builtin_amdgcn_s_wait_asynccnt)
  __builtin_amdgcn_s_wait_asynccnt(0);
#else
  asm volatile("s_wait_asynccnt 0x0" ::: "memory");
#endif
}
#endif

// ---------------- generic batched WMMA GEMM ----------------
// C[z] = epilogue( A[z] (MxK, bf16) * B[z] (KxN or N-major, bf16) )
// batch offsets: offA=(z%aMod)*sA1+(z/aMod)*sA2 ; B uses zb=bIdxB?bIdxB[z]:z
#define GF_GELU    1
#define GF_STOREBF 2
#define GF_ATOMIC  4

struct GemmP {
  const bf16_t* A; const bf16_t* B; void* C;
  const float* bias; const float* residual; const float* wscale; const int* bIdxB;
  int M, N, Kd, lda, ldb, ldc;
  int aMod; long long sA1, sA2;
  int bMod; long long sB1, sB2;
  int cMod; long long sC1, sC2;
  long long biasStride;
  int opB;            // 0: B is Kd x N row-major ; 1: B given transposed (N x Kd)
  int flags; float alpha;
};

#define BM 128
#define BN 128
#define BK 32
#define LDK 40   // padded K stride in LDS (bf16 elems)

__launch_bounds__(256)
__global__ void gemm_wmma(GemmP p) {
  __shared__ bf16_t As[BM * LDK];
  __shared__ bf16_t Bs[BN * LDK];
  const int tid  = threadIdx.x;
  const int lane = tid & 31;
  const int wave = tid >> 5;         // 0..7
  const int wm   = wave >> 1;        // 0..3 : 32-row stripe
  const int wn   = wave & 1;         // 0..1 : 64-col stripe
  const int z    = blockIdx.z;
  const int m0   = blockIdx.y * BM;
  const int n0   = blockIdx.x * BN;

  const int zb = p.bIdxB ? p.bIdxB[z] : z;
  const bf16_t* Ag = p.A + (long long)(z % p.aMod) * p.sA1 + (long long)(z / p.aMod) * p.sA2;
  const bf16_t* Bg = p.B + (long long)(zb % p.bMod) * p.sB1 + (long long)(zb / p.bMod) * p.sB2;
  const long long offC = (long long)(z % p.cMod) * p.sC1 + (long long)(z / p.cMod) * p.sC2;
  const float* biasp = p.bias ? (p.bias + (long long)zb * p.biasStride) : nullptr;

  v8f acc[2][4];
#pragma unroll
  for (int mt = 0; mt < 2; ++mt)
#pragma unroll
    for (int nt = 0; nt < 4; ++nt)
#pragma unroll
      for (int r = 0; r < 8; ++r) acc[mt][nt][r] = 0.0f;

  for (int k0 = 0; k0 < p.Kd; k0 += BK) {
    // ---- stage A tile (BM x BK) ----
#pragma unroll
    for (int rr = 0; rr < 2; ++rr) {
      int id = tid + rr * 256;
      int row = id >> 2, cc = (id & 3) * 8;
      int gm = m0 + row;
      const bf16_t* gp = Ag + (long long)gm * p.lda + (k0 + cc);
      bf16_t* lp = &As[row * LDK + cc];
#ifdef HAVE_ASYNC_LDS
      if (gm < p.M) {
        async_cp16(gp, lp);
        if (k0 + BK < p.Kd) __builtin_prefetch(gp + BK, 0, 2);
      } else {
        *(v8bf*)lp = zero8();
      }
#else
      v8bf v = zero8();
      if (gm < p.M) v = *(const v8bf*)gp;
      *(v8bf*)lp = v;
#endif
    }
    // ---- stage B tile as N-major (BN x BK) ----
    if (p.opB) {
#pragma unroll
      for (int rr = 0; rr < 2; ++rr) {
        int id = tid + rr * 256;
        int row = id >> 2, cc = (id & 3) * 8;
        int gn = n0 + row;
        const bf16_t* gp = Bg + (long long)gn * p.ldb + (k0 + cc);
        bf16_t* lp = &Bs[row * LDK + cc];
#ifdef HAVE_ASYNC_LDS
        if (gn < p.N) {
          async_cp16(gp, lp);
          if (k0 + BK < p.Kd) __builtin_prefetch(gp + BK, 0, 2);
        } else {
          *(v8bf*)lp = zero8();
        }
#else
        v8bf v = zero8();
        if (gn < p.N) v = *(const v8bf*)gp;
        *(v8bf*)lp = v;
#endif
      }
    } else {
      // transpose during staging (K-major global -> N-major LDS)
#pragma unroll
      for (int rr = 0; rr < 2; ++rr) {
        int id = tid + rr * 256;
        int kk = id >> 4, cn = (id & 15) * 8;
        const bf16_t* gp = Bg + (long long)(k0 + kk) * p.ldb + (n0 + cn);
        v8bf v = zero8();
        if (n0 + cn < p.N) {
          v = *(const v8bf*)gp;
          if (k0 + BK < p.Kd) __builtin_prefetch(gp + (long long)BK * p.ldb, 0, 2);
        }
#pragma unroll
        for (int j = 0; j < 8; ++j) Bs[(cn + j) * LDK + kk] = v[j];
      }
    }
#ifdef HAVE_ASYNC_LDS
    wait_async0();
#endif
    __syncthreads();

    // ---- fragments per ISA 7.12.2 layout, then 8 WMMAs ----
    const int ra   = lane & 15;
    const int kbA  = (lane >= 16) ? 8 : 0;
    const int kbB  = (lane >= 16) ? 8 : 0;
    v16bf afr[2], bfr[4];
#pragma unroll
    for (int mt = 0; mt < 2; ++mt) {
      int row = wm * 32 + mt * 16 + ra;
      v8bf h0 = *(const v8bf*)(&As[row * LDK + kbA]);
      v8bf h1 = *(const v8bf*)(&As[row * LDK + kbA + 16]);
      afr[mt] = __builtin_shufflevector(h0, h1, 0,1,2,3,4,5,6,7,8,9,10,11,12,13,14,15);
    }
#pragma unroll
    for (int nt = 0; nt < 4; ++nt) {
      int col = wn * 64 + nt * 16 + ra;
      v8bf h0 = *(const v8bf*)(&Bs[col * LDK + kbB]);
      v8bf h1 = *(const v8bf*)(&Bs[col * LDK + kbB + 16]);
      bfr[nt] = __builtin_shufflevector(h0, h1, 0,1,2,3,4,5,6,7,8,9,10,11,12,13,14,15);
    }
#pragma unroll
    for (int mt = 0; mt < 2; ++mt)
#pragma unroll
      for (int nt = 0; nt < 4; ++nt)
        acc[mt][nt] = __builtin_amdgcn_wmma_f32_16x16x32_bf16(
            false, afr[mt], false, bfr[nt], (short)0, acc[mt][nt], false, false);
    __syncthreads();
  }

  // ---- epilogue ----
  const float s = p.wscale ? p.wscale[z] : 1.0f;
#pragma unroll
  for (int mt = 0; mt < 2; ++mt) {
#pragma unroll
    for (int nt = 0; nt < 4; ++nt) {
      int gn = n0 + wn * 64 + nt * 16 + (lane & 15);
      int gmB = m0 + wm * 32 + mt * 16 + ((lane >> 4) << 3);
      if (gn < p.N) {
        float bi = biasp ? biasp[gn] : 0.0f;
#pragma unroll
        for (int r = 0; r < 8; ++r) {
          int gm = gmB + r;
          if (gm < p.M) {
            float v = acc[mt][nt][r] + bi;
            if (p.flags & GF_GELU) v = gelu_f(v);
            long long ci = offC + (long long)gm * p.ldc + gn;
            if (p.residual) v += p.residual[ci];
            v *= s * p.alpha;
            if (p.flags & GF_ATOMIC)       atomicAdd((float*)p.C + ci, v);
            else if (p.flags & GF_STOREBF) ((bf16_t*)p.C)[ci] = f2bf(v);
            else                           ((float*)p.C)[ci]  = v;
          }
        }
      }
    }
  }
}

// ---------------- fused 3-output LayerNorm -> bf16 ----------------
__launch_bounds__(256)
__global__ void ln3_kernel(const float* X,
                           const float* w0, const float* b0, bf16_t* o0,
                           const float* w1, const float* b1, bf16_t* o1,
                           const float* w2, const float* b2, bf16_t* o2) {
  __shared__ float red[256];
  const long long row = blockIdx.x;
  const float* xr = X + row * DIM;
  const int t = threadIdx.x;
  float vals[4]; float lsum = 0.f;
#pragma unroll
  for (int i = 0; i < 4; ++i) { vals[i] = xr[t + i * 256]; lsum += vals[i]; }
  red[t] = lsum; __syncthreads();
  for (int s = 128; s > 0; s >>= 1) { if (t < s) red[t] += red[t + s]; __syncthreads(); }
  float mu = red[0] * (1.0f / DIM); __syncthreads();
  float lvs = 0.f;
#pragma unroll
  for (int i = 0; i < 4; ++i) { float d = vals[i] - mu; lvs += d * d; }
  red[t] = lvs; __syncthreads();
  for (int s = 128; s > 0; s >>= 1) { if (t < s) red[t] += red[t + s]; __syncthreads(); }
  float rstd = rsqrtf(red[0] * (1.0f / DIM) + EPSV);
#pragma unroll
  for (int i = 0; i < 4; ++i) {
    int c = t + i * 256;
    float xn = (vals[i] - mu) * rstd;
    o0[row * DIM + c] = f2bf(xn * w0[c] + b0[c]);
    if (o1) o1[row * DIM + c] = f2bf(xn * w1[c] + b1[c]);
    if (o2) o2[row * DIM + c] = f2bf(xn * w2[c] + b2[c]);
  }
}

// ---------------- row softmax (512 cols) -> bf16 ----------------
__launch_bounds__(256)
__global__ void softmax_kernel(const float* Sc, bf16_t* P, float alpha) {
  __shared__ float red[256];
  const long long row = blockIdx.x;
  const int t = threadIdx.x;
  const float* sr = Sc + row * SEQ;
  float v0 = sr[t] * alpha, v1 = sr[t + 256] * alpha;
  red[t] = fmaxf(v0, v1); __syncthreads();
  for (int s = 128; s > 0; s >>= 1) { if (t < s) red[t] = fmaxf(red[t], red[t + s]); __syncthreads(); }
  float mx = red[0]; __syncthreads();
  float e0 = __expf(v0 - mx), e1 = __expf(v1 - mx);
  red[t] = e0 + e1; __syncthreads();
  for (int s = 128; s > 0; s >>= 1) { if (t < s) red[t] += red[t + s]; __syncthreads(); }
  float inv = 1.0f / red[0];
  P[row * SEQ + t]       = f2bf(e0 * inv);
  P[row * SEQ + t + 256] = f2bf(e1 * inv);
}

// ---------------- router-input mean over LR ----------------
__launch_bounds__(256)
__global__ void rimean_kernel(const float* RI, float* out) {
  const int b = blockIdx.x, t = threadIdx.x;
#pragma unroll
  for (int j = 0; j < 4; ++j) {
    int c = t + j * 256;
    float s = 0.f;
    for (int r = 0; r < LROUT; ++r) s += RI[(long long)b * LROUT * DIM + (long long)r * DIM + c];
    out[b * DIM + c] = s * (1.0f / LROUT);
  }
}

// ---------------- f32 -> bf16 conversion ----------------
__launch_bounds__(256)
__global__ void cvt_kernel(const float* in, bf16_t* out, long long n) {
  long long i = (long long)blockIdx.x * 256 + threadIdx.x;
  long long stride = (long long)gridDim.x * 256;
  for (; i < n; i += stride) out[i] = f2bf(in[i]);
}

// ---------------- noisy top-2 router + capacity ----------------
__device__ inline unsigned pcg_h(unsigned v) {
  unsigned s = v * 747796405u + 2891336453u;
  unsigned w = ((s >> ((s >> 28) + 4u)) ^ s) * 277803737u;
  return (w >> 22) ^ w;
}
__device__ inline float nrand_h(int l, int b, int e) {
  unsigned s = (unsigned)(l * 0x9E3779B1u) ^ (unsigned)(b * 0x85EBCA77u) ^
               (unsigned)(e * 0xC2B2AE3Du) ^ 0x27220A95u;
  unsigned a = pcg_h(s), c = pcg_h(s ^ 0x6A09E667u);
  float u1 = ((a >> 8) + 0.5f) * (1.0f / 16777216.0f);
  float u2 = ((c >> 8) + 0.5f) * (1.0f / 16777216.0f);
  return sqrtf(-2.0f * logf(u1)) * cosf(6.2831853f * u2);
}
__launch_bounds__(64)
__global__ void router_kernel(const float* rim, const float* Wr, const float* br,
                              const float* Wn, const float* bn, int layer,
                              int* flat_e, float* flat_w) {
  __shared__ float lg[NB][NEXP], nl[NB][NEXP];
  __shared__ int   ti[NB][TOPK];
  __shared__ float gw[NB][TOPK];
  const int t = threadIdx.x, b = t >> 3, e = t & 7;
  float s1 = 0.f, s2 = 0.f;
  for (int d = 0; d < DIM; ++d) {
    float x = rim[b * DIM + d];
    s1 += x * Wr[(long long)d * NEXP + e];
    s2 += x * Wn[(long long)d * NEXP + e];
  }
  lg[b][e] = s1 + br[e]; nl[b][e] = s2 + bn[e];
  __syncthreads();
  if (t < NB) {
    float v[NEXP];
    for (int ee = 0; ee < NEXP; ++ee) {
      float x = nl[t][ee];
      float sp = (x > 20.f) ? x : log1pf(__expf(x));
      v[ee] = lg[t][ee] + nrand_h(layer, t, ee) * sp;
    }
    int i1 = 0;
    for (int ee = 1; ee < NEXP; ++ee) if (v[ee] > v[i1]) i1 = ee;
    int i2 = (i1 == 0) ? 1 : 0;
    for (int ee = 0; ee < NEXP; ++ee) if (ee != i1 && v[ee] > v[i2]) i2 = ee;
    float m = fmaxf(v[i1], v[i2]);
    float e1 = __expf(v[i1] - m), e2 = __expf(v[i2] - m);
    float inv = 1.0f / (e1 + e2);
    ti[t][0] = i1; ti[t][1] = i2; gw[t][0] = e1 * inv; gw[t][1] = e2 * inv;
  }
  __syncthreads();
  if (t == 0) {
    int cnt[NEXP];
    for (int ee = 0; ee < NEXP; ++ee) cnt[ee] = 0;
    for (int n = 0; n < NB * TOPK; ++n) {
      int bb = n >> 1, j = n & 1, ee = ti[bb][j];
      int rank = cnt[ee]++;
      flat_e[n] = ee;
      flat_w[n] = (rank < CAPV) ? gw[bb][j] : 0.0f;
    }
  }
}

// ---------------- host side ----------------
static void launch_gemm(hipStream_t st, const bf16_t* A, const bf16_t* B, void* C,
                        const float* bias, const float* residual, const float* wscale,
                        const int* bIdxB, int M, int N, int K, int lda, int ldb, int ldc,
                        int aMod, long long sA1, long long sA2,
                        int bMod, long long sB1, long long sB2,
                        int cMod, long long sC1, long long sC2,
                        long long biasStride, int opB, int flags, float alpha, int batch) {
  GemmP p;
  p.A = A; p.B = B; p.C = C; p.bias = bias; p.residual = residual;
  p.wscale = wscale; p.bIdxB = bIdxB;
  p.M = M; p.N = N; p.Kd = K; p.lda = lda; p.ldb = ldb; p.ldc = ldc;
  p.aMod = aMod; p.sA1 = sA1; p.sA2 = sA2;
  p.bMod = bMod; p.sB1 = sB1; p.sB2 = sB2;
  p.cMod = cMod; p.sC1 = sC1; p.sC2 = sC2;
  p.biasStride = biasStride; p.opB = opB; p.flags = flags; p.alpha = alpha;
  dim3 grid((N + BN - 1) / BN, (M + BM - 1) / BM, batch);
  gemm_wmma<<<grid, 256, 0, st>>>(p);
}

extern "C" void kernel_launch(void* const* d_in, const int* in_sizes, int n_in,
                              void* d_out, int out_size, void* d_ws, size_t ws_size,
                              hipStream_t stream) {
  (void)in_sizes; (void)n_in; (void)ws_size;
  const float* RI   = (const float*)d_in[0];
  const float* Xin  = (const float*)d_in[1];
  const float* nqw  = (const float*)d_in[2];
  const float* nqb  = (const float*)d_in[3];
  const float* nkw  = (const float*)d_in[4];
  const float* nkb  = (const float*)d_in[5];
  const float* nvw  = (const float*)d_in[6];
  const float* nvb  = (const float*)d_in[7];
  const float* Wq   = (const float*)d_in[8];
  const float* Wk   = (const float*)d_in[9];
  const float* Wv   = (const float*)d_in[10];
  const float* Wo   = (const float*)d_in[11];
  const float* bo   = (const float*)d_in[12];
  const float* nmw  = (const float*)d_in[13];
  const float* nmb  = (const float*)d_in[14];
  const float* Wr   = (const float*)d_in[15];
  const float* br   = (const float*)d_in[16];
  const float* Wn   = (const float*)d_in[17];
  const float* bn   = (const float*)d_in[18];
  const float* W1   = (const float*)d_in[19];
  const float* b1   = (const float*)d_in[20];
  const float* W2   = (const float*)d_in[21];
  const float* b2   = (const float*)d_in[22];

  const long long ROWS = (long long)NB * SEQ;              // 4096
  const long long NSLOT = NB * TOPK;                       // 16

  char* ws = (char*)d_ws;
  size_t off = 0;
  auto alloc = [&](size_t bytes) -> char* {
    char* p = ws + off; off += (bytes + 255) & ~(size_t)255; return p;
  };
  float*  X    = (float*) alloc(ROWS * DIM * 4);                       // 16 MB
  bf16_t* WqB  = (bf16_t*)alloc((size_t)NLAYER * DIM * INNER * 2);
  bf16_t* WkB  = (bf16_t*)alloc((size_t)NLAYER * DIM * INNER * 2);
  bf16_t* WvB  = (bf16_t*)alloc((size_t)NLAYER * DIM * INNER * 2);
  bf16_t* WoB  = (bf16_t*)alloc((size_t)NLAYER * INNER * DIM * 2);
  bf16_t* W1B  = (bf16_t*)alloc((size_t)NLAYER * NEXP * DIM * MLPD * 2);   // 128 MB
  bf16_t* W2B  = (bf16_t*)alloc((size_t)NLAYER * NEXP * MLPD * DIM * 2);   // 128 MB
  bf16_t* XnQ  = (bf16_t*)alloc(ROWS * DIM * 2);
  bf16_t* XnK  = (bf16_t*)alloc(ROWS * DIM * 2);
  bf16_t* XnV  = (bf16_t*)alloc(ROWS * DIM * 2);
  bf16_t* Qb   = (bf16_t*)alloc(ROWS * DIM * 2);
  bf16_t* Kb   = (bf16_t*)alloc(ROWS * DIM * 2);
  bf16_t* Vb   = (bf16_t*)alloc(ROWS * DIM * 2);
  float*  Sc   = (float*) alloc((size_t)NB * NHEADS * SEQ * SEQ * 4);  // 128 MB
  bf16_t* Pb   = (bf16_t*)alloc((size_t)NB * NHEADS * SEQ * SEQ * 2);  // 64 MB
  bf16_t* Ob   = (bf16_t*)alloc(ROWS * DIM * 2);
  float*  riM  = (float*) alloc(NB * DIM * 4);
  int*    flE  = (int*)   alloc(NSLOT * 4);
  float*  flW  = (float*) alloc(NSLOT * 4);
  bf16_t* Xm   = XnQ;            // lifetime reuse (XnQ dead after QKV GEMMs)
  bf16_t* H    = (bf16_t*)Sc;    // lifetime reuse (scores dead after softmax)

  // init x working buffer
  (void)hipMemcpyAsync(X, Xin, (size_t)ROWS * DIM * 4, hipMemcpyDeviceToDevice, stream);

  // weights -> bf16 (one-time per launch)
  cvt_kernel<<<2048, 256, 0, stream>>>(Wq, WqB, (long long)NLAYER * DIM * INNER);
  cvt_kernel<<<2048, 256, 0, stream>>>(Wk, WkB, (long long)NLAYER * DIM * INNER);
  cvt_kernel<<<2048, 256, 0, stream>>>(Wv, WvB, (long long)NLAYER * DIM * INNER);
  cvt_kernel<<<2048, 256, 0, stream>>>(Wo, WoB, (long long)NLAYER * INNER * DIM);
  cvt_kernel<<<4096, 256, 0, stream>>>(W1, W1B, (long long)NLAYER * NEXP * DIM * MLPD);
  cvt_kernel<<<4096, 256, 0, stream>>>(W2, W2B, (long long)NLAYER * NEXP * MLPD * DIM);

  rimean_kernel<<<NB, 256, 0, stream>>>(RI, riM);

  const float scale = 1.0f / sqrtf((float)DHEAD);

  for (int l = 0; l < NLAYER; ++l) {
    // --- pre-norm x3 ---
    ln3_kernel<<<(unsigned)ROWS, 256, 0, stream>>>(X,
        nqw + l * DIM, nqb + l * DIM, XnQ,
        nkw + l * DIM, nkb + l * DIM, XnK,
        nvw + l * DIM, nvb + l * DIM, XnV);

    // --- QKV projections: [4096x1024] x [1024x1024] -> bf16 ---
    launch_gemm(stream, XnQ, WqB + (long long)l * DIM * INNER, Qb, nullptr, nullptr, nullptr, nullptr,
                (int)ROWS, INNER, DIM, DIM, INNER, INNER,
                1, 0, 0,  1, 0, 0,  1, 0, 0,  0, 0, GF_STOREBF, 1.0f, 1);
    launch_gemm(stream, XnK, WkB + (long long)l * DIM * INNER, Kb, nullptr, nullptr, nullptr, nullptr,
                (int)ROWS, INNER, DIM, DIM, INNER, INNER,
                1, 0, 0,  1, 0, 0,  1, 0, 0,  0, 0, GF_STOREBF, 1.0f, 1);
    launch_gemm(stream, XnV, WvB + (long long)l * DIM * INNER, Vb, nullptr, nullptr, nullptr, nullptr,
                (int)ROWS, INNER, DIM, DIM, INNER, INNER,
                1, 0, 0,  1, 0, 0,  1, 0, 0,  0, 0, GF_STOREBF, 1.0f, 1);

    // --- attention scores: per (b,h)  Q[512x64] * K^T -> f32 ---
    launch_gemm(stream, Qb, Kb, Sc, nullptr, nullptr, nullptr, nullptr,
                SEQ, SEQ, DHEAD, DIM, DIM, SEQ,
                NHEADS, DHEAD, (long long)SEQ * DIM,
                NHEADS, DHEAD, (long long)SEQ * DIM,
                1, 0, (long long)SEQ * SEQ,
                0, 1 /*opB=T*/, 0, 1.0f, NB * NHEADS);

    // --- softmax rows (scale folded in) -> bf16 P ---
    softmax_kernel<<<(unsigned)((long long)NB * NHEADS * SEQ), 256, 0, stream>>>(Sc, Pb, scale);

    // --- P*V: per (b,h)  [512x512] x [512x64] -> Ob bf16 (packed as [B,S,INNER]) ---
    launch_gemm(stream, Pb, Vb, Ob, nullptr, nullptr, nullptr, nullptr,
                SEQ, DHEAD, SEQ, SEQ, DIM, DIM,
                1, 0, (long long)SEQ * SEQ,
                NHEADS, DHEAD, (long long)SEQ * DIM,
                NHEADS, DHEAD, (long long)SEQ * DIM,
                0, 0, GF_STOREBF, 1.0f, NB * NHEADS);

    // --- output projection + bias + residual: X = X + Ob*Wo + bo ---
    launch_gemm(stream, Ob, WoB + (long long)l * INNER * DIM, X, bo + l * DIM, X, nullptr, nullptr,
                (int)ROWS, DIM, INNER, DIM, DIM, DIM,
                1, 0, 0,  1, 0, 0,  1, 0, 0,  0, 0, 0 /*f32 store*/, 1.0f, 1);

    // --- noisy top-2 routing + capacity ---
    router_kernel<<<1, 64, 0, stream>>>(riM,
        Wr + (long long)l * DIM * NEXP, br + l * NEXP,
        Wn + (long long)l * DIM * NEXP, bn + l * NEXP, l, flE, flW);

    // --- MoE pre-norm ---
    ln3_kernel<<<(unsigned)ROWS, 256, 0, stream>>>(X,
        nmw + l * DIM, nmb + l * DIM, Xm,
        nullptr, nullptr, nullptr, nullptr, nullptr, nullptr);

    // --- expert FFN 1: H[z] = gelu(Xm[z/2] * W1[e(z)] + b1[e(z)]) -> bf16 ---
    launch_gemm(stream, Xm, W1B + (long long)l * NEXP * DIM * MLPD, H,
                b1 + (long long)l * NEXP * MLPD, nullptr, nullptr, flE,
                SEQ, MLPD, DIM, DIM, MLPD, MLPD,
                TOPK, 0, (long long)SEQ * DIM,
                1, 0, (long long)DIM * MLPD,
                1, 0, (long long)SEQ * MLPD,
                MLPD, 0, GF_GELU | GF_STOREBF, 1.0f, (int)NSLOT);

    // --- expert FFN 2 + gate scale + scatter-combine: X += w[z]*(H[z]*W2[e(z)] + b2[e(z)]) ---
    launch_gemm(stream, H, W2B + (long long)l * NEXP * MLPD * DIM, X,
                b2 + (long long)l * NEXP * DIM, nullptr, flW, flE,
                SEQ, DIM, MLPD, MLPD, DIM, DIM,
                1, 0, (long long)SEQ * MLPD,
                1, 0, (long long)MLPD * DIM,
                TOPK, 0, (long long)SEQ * DIM,
                DIM, 0, GF_ATOMIC, 1.0f, (int)NSLOT);
  }

  (void)hipMemcpyAsync(d_out, X, (size_t)out_size * sizeof(float), hipMemcpyDeviceToDevice, stream);
}
